// GPTModel_45234595562047
// MI455X (gfx1250) — matrix operand order repaired and checked by reading, not compiled, not run on hardware
//
#include <hip/hip_runtime.h>
#include <hip/hip_bf16.h>
#include <math.h>

// ---------------------------------------------------------------------------
// GPT forward for MI455X (gfx1250, wave32, WMMA 16x16x32 f16, f32 accum,
// TDM async tensor loads for A tiles)
// ---------------------------------------------------------------------------

typedef unsigned short u16;
typedef _Float16 v16h __attribute__((ext_vector_type(16)));
typedef float    v8f  __attribute__((ext_vector_type(8)));
typedef unsigned int u32x4 __attribute__((ext_vector_type(4)));
typedef int          i32x8 __attribute__((ext_vector_type(8)));
typedef int          i32x4 __attribute__((ext_vector_type(4)));

#define GPT_B  2
#define GPT_T  1024
#define GPT_D  1024
#define GPT_H  16
#define GPT_HS 64
#define GPT_L  8
#define GPT_V  32000
#define GPT_M  (GPT_B * GPT_T)      // 2048 token rows
#define GPT_FF (4 * GPT_D)          // 4096

#if defined(__has_builtin)
#if __has_builtin(__builtin_amdgcn_tensor_load_to_lds) && \
    __has_builtin(__builtin_amdgcn_s_wait_tensorcnt)
#define USE_TDM 1
#endif
#endif

union AFrag { v16h v; uint4 q[2]; u16 h[16]; };
union CFrag { v8f  v; float f[8]; };

__device__ __forceinline__ u16 f2h(float f) {
    union { _Float16 h; u16 u; } c;
    c.h = (_Float16)f;
    return c.u;
}

#ifdef USE_TDM
// 2D TDM load: tile_rows x tile_k f16 elements, row stride `row_stride`
// elements in global memory, packed row-major into LDS at byte offset lds_off.
// D# packing per CDNA5 ISA section 8.3/8.4 (tensor dims == tile dims -> no OOB).
// This toolchain exposes the 6-arg builtin: (g0, g1, g2, g3, g4, cpol).
__device__ __forceinline__ void tdm_load_tile_f16(
    unsigned lds_off, const void* gptr,
    unsigned tile_k, unsigned tile_rows, unsigned row_stride)
{
    const unsigned long long ga = (unsigned long long)gptr;
    u32x4 g0;
    g0[0] = 1u;                                   // count=1, user descriptor
    g0[1] = lds_off;                              // lds_addr (bytes)
    g0[2] = (unsigned)(ga & 0xffffffffu);         // global_addr[31:0]
    g0[3] = (unsigned)((ga >> 32) & 0x01ffffffu)  // global_addr[56:32]
          | (2u << 30);                           // type = 2 ("image")
    i32x8 g1;
    g1[0] = (int)(1u << 16);                      // data_size = 1 -> 2 bytes
    g1[1] = (int)((tile_k & 0xffffu) << 16);      // tensor_dim0[15:0]
    g1[2] = (int)(((tile_k >> 16) & 0xffffu) | ((tile_rows & 0xffffu) << 16));
    g1[3] = (int)(((tile_rows >> 16) & 0xffffu) | ((tile_k & 0xffffu) << 16)); // tile_dim0
    g1[4] = (int)(tile_rows & 0xffffu);           // tile_dim1 (tile_dim2 = 0)
    g1[5] = (int)row_stride;                      // tensor_dim0_stride[31:0]
    g1[6] = 0;                                    // stride[47:32] | dim1_stride lo
    g1[7] = 0;
    i32x4 z4;
    z4[0] = 0; z4[1] = 0; z4[2] = 0; z4[3] = 0;
    i32x8 z8;
    z8[0] = 0; z8[1] = 0; z8[2] = 0; z8[3] = 0;
    z8[4] = 0; z8[5] = 0; z8[6] = 0; z8[7] = 0;
    __builtin_amdgcn_tensor_load_to_lds(g0, g1, z4, z4, z8, 0);
}
#endif

// ---------------------------------------------------------------------------
// Embedding + sinusoidal positional encoding (fp32 residual stream)
// ---------------------------------------------------------------------------
__global__ __launch_bounds__(256) void embed_kernel(
    const int* __restrict__ idx, const float* __restrict__ tok,
    float* __restrict__ x)
{
    const int row = blockIdx.x;              // b*T + t
    const int t   = row & (GPT_T - 1);
    const int token = idx[row];
    const float c0 = -logf(10000.0f) / (float)GPT_D;
#pragma unroll
    for (int i = 0; i < 4; ++i) {
        const int d  = threadIdx.x + i * 256;
        const int j2 = d & ~1;               // 2*j (even index of pair)
        const float ang = (float)t * expf((float)j2 * c0);
        const float pe  = (d & 1) ? cosf(ang) : sinf(ang);
        x[(long)row * GPT_D + d] = tok[(long)token * GPT_D + d] + pe;
    }
}

// ---------------------------------------------------------------------------
// LayerNorm: fp32 in -> f16 out (A operand for subsequent GEMMs)
// ---------------------------------------------------------------------------
__global__ __launch_bounds__(256) void ln_kernel(
    const float* __restrict__ x, const float* __restrict__ gamma,
    const float* __restrict__ beta, u16* __restrict__ out)
{
    __shared__ float red[256];
    const int tid = threadIdx.x;
    const long row = blockIdx.x;
    const float* xr = x + row * GPT_D;

    float v[4];
    float s = 0.f;
#pragma unroll
    for (int i = 0; i < 4; ++i) { v[i] = xr[tid + i * 256]; s += v[i]; }
    red[tid] = s; __syncthreads();
    for (int off = 128; off > 0; off >>= 1) {
        if (tid < off) red[tid] += red[tid + off];
        __syncthreads();
    }
    const float mean = red[0] * (1.f / GPT_D);
    __syncthreads();

    float s2 = 0.f;
#pragma unroll
    for (int i = 0; i < 4; ++i) { float d = v[i] - mean; s2 += d * d; }
    red[tid] = s2; __syncthreads();
    for (int off = 128; off > 0; off >>= 1) {
        if (tid < off) red[tid] += red[tid + off];
        __syncthreads();
    }
    const float rstd = rsqrtf(red[0] * (1.f / GPT_D) + 1e-5f);

    u16* orow = out + row * GPT_D;
#pragma unroll
    for (int i = 0; i < 4; ++i) {
        const int c = tid + i * 256;
        orow[c] = f2h((v[i] - mean) * rstd * gamma[c] + beta[c]);
    }
}

// ---------------------------------------------------------------------------
// Tiled WMMA GEMM: C[M,N] = A[M,K](f16) x B[K,N](f32 weights, converted), with
// fused bias / relu / fp32 residual.  Block tile 128 x TN x 32, 256 threads.
//   TN=64 : 8 waves stacked in M, wave tile 16x64 (4 wmma/k-step)  [QKV heads]
//   TN=128: 4x2 wave grid, wave tile 32x64 (8 wmma/k-step)         [big GEMMs]
// A tile staged via Tensor Data Mover (async DMA, TENSORcnt) when available.
//   BTRANS: B element (k,n) read from Bsrc[n*ldb + k]  (tied-embedding logits)
// ---------------------------------------------------------------------------
template <int TN, bool BTRANS, bool BIAS, bool RELU, bool RESID, bool STF, bool STB>
__global__ __launch_bounds__(256) void gemm_kernel(
    const u16*  __restrict__ A,  int lda,
    const float* __restrict__ Bsrc, int ldb, long b_zstride,
    const float* __restrict__ bias,
    const float* __restrict__ resid, int ldr,
    float* __restrict__ Cf, int ldcf,
    u16*   __restrict__ Cb, int ldcb,
    int K, int c_zcol)
{
    constexpr int WN = TN / 64;       // waves along N (1 or 2)
    constexpr int WM = 8 / WN;        // waves along M (8 or 4)
    constexpr int MF = 128 / (16 * WM); // 16-row frags per wave (1 or 2)

    __shared__ u16 As[128][32];       // 8 KB, row-major [m][k]
    __shared__ u16 Bs[TN][32];        // [n][k] (K contiguous per column)

    const int tid  = threadIdx.x;
    const int wave = tid >> 5;
    const int lane = tid & 31;
    const int half = lane >> 4;
    const int l16  = lane & 15;
    const int wm   = wave % WM;
    const int wn   = wave / WM;
    const int mbase = blockIdx.y * 128;
    const int nbase = blockIdx.x * TN;
    const float* Bp = Bsrc + (long)blockIdx.z * b_zstride;

    CFrag acc[MF][4];
#pragma unroll
    for (int mf = 0; mf < MF; ++mf)
#pragma unroll
        for (int n = 0; n < 4; ++n)
#pragma unroll
            for (int r = 0; r < 8; ++r) acc[mf][n].f[r] = 0.f;

#ifdef USE_TDM
    // generic LDS pointer low 32 bits == LDS byte offset (ISA 10.2 aperture map)
    const unsigned as_off = (unsigned)(unsigned long long)(const void*)&As[0][0];
#endif

    for (int kb = 0; kb < K; kb += 32) {
#ifdef USE_TDM
        // A tile (128x32 f16) via Tensor Data Mover: wave 0 issues one DMA.
        if (tid < 32) {
            tdm_load_tile_f16(as_off, &A[(long)mbase * lda + kb],
                              32u, 128u, (unsigned)lda);
        }
#else
#pragma unroll
        for (int i = 0; i < 2; ++i) {
            const int li  = tid + i * 256;
            const int row = li >> 2;
            const int cc  = li & 3;
            *(uint4*)&As[row][cc * 8] =
                *(const uint4*)&A[(long)(mbase + row) * lda + kb + cc * 8];
        }
#endif
        // stage B tile (f32 -> f16, into [n][k]); overlaps with the A DMA
        if (!BTRANS) {
#pragma unroll
            for (int i = 0; i < TN / 32; ++i) {
                const int li  = tid + i * 256;     // float4 index
                const int row = li / (TN / 4);     // k within tile
                const int cc  = li % (TN / 4);     // 4-col chunk
                float4 f = *(const float4*)&Bp[(long)(kb + row) * ldb + nbase + cc * 4];
                Bs[cc * 4 + 0][row] = f2h(f.x);
                Bs[cc * 4 + 1][row] = f2h(f.y);
                Bs[cc * 4 + 2][row] = f2h(f.z);
                Bs[cc * 4 + 3][row] = f2h(f.w);
                if (kb + 32 < K)
                    __builtin_prefetch(&Bp[(long)(kb + 32 + row) * ldb + nbase + cc * 4], 0, 1);
            }
        } else {
#pragma unroll
            for (int i = 0; i < TN / 32; ++i) {
                const int li  = tid + i * 256;
                const int col = li >> 3;           // n within tile
                const int kc  = li & 7;            // 4-k chunk
                float4 f = *(const float4*)&Bp[(long)(nbase + col) * ldb + kb + kc * 4];
                Bs[col][kc * 4 + 0] = f2h(f.x);
                Bs[col][kc * 4 + 1] = f2h(f.y);
                Bs[col][kc * 4 + 2] = f2h(f.z);
                Bs[col][kc * 4 + 3] = f2h(f.w);
                if (kb + 32 < K)
                    __builtin_prefetch(&Bp[(long)(nbase + col) * ldb + kb + 32 + kc * 4], 0, 1);
            }
        }
#ifdef USE_TDM
        if (tid < 32) __builtin_amdgcn_s_wait_tensorcnt(0);
#endif
        __syncthreads();

        // A fragments: lane = M row; k = {half*8+0..7, 16+half*8+0..7}
        AFrag a[MF];
#pragma unroll
        for (int mf = 0; mf < MF; ++mf) {
            const u16* ap = &As[(wm * MF + mf) * 16 + l16][half * 8];
            a[mf].q[0] = *(const uint4*)ap;
            a[mf].q[1] = *(const uint4*)(ap + 16);
        }
#pragma unroll
        for (int n = 0; n < 4; ++n) {
            // B fragment: lane col = l16, k = {half*8+0..7, 16+half*8+0..7}
            AFrag bfr;
            const u16* bp = &Bs[wn * 64 + n * 16 + l16][half * 8];
            bfr.q[0] = *(const uint4*)bp;
            bfr.q[1] = *(const uint4*)(bp + 16);
#pragma unroll
            for (int mf = 0; mf < MF; ++mf)
                acc[mf][n].v = __builtin_amdgcn_wmma_f32_16x16x32_f16(
                    false, a[mf].v, false, bfr.v, (short)0, acc[mf][n].v, false, false);
        }
        __syncthreads();
    }

    // epilogue: C/D layout -> row = r + 8*half, col = l16 within 16-col subtile
    const int gcolbase = blockIdx.z * c_zcol + nbase + wn * 64;
#pragma unroll
    for (int mf = 0; mf < MF; ++mf) {
#pragma unroll
        for (int n = 0; n < 4; ++n) {
            const int gc = gcolbase + n * 16 + l16;
#pragma unroll
            for (int r = 0; r < 8; ++r) {
                const int gr = mbase + (wm * MF + mf) * 16 + r + 8 * half;
                float val = acc[mf][n].f[r];
                if (BIAS)  val += bias[gc];
                if (RELU)  val = fmaxf(val, 0.f);
                if (RESID) val += resid[(long)gr * ldr + gc];
                if (STF) Cf[(long)gr * ldcf + gc] = val;
                if (STB) Cb[(long)gr * ldcb + gc] = f2h(val);
            }
        }
    }
}

// ---------------------------------------------------------------------------
// Flash attention: grid (T/64, H, B), 128 threads = 4 waves. Each wave owns a
// 16-row query tile; keys processed 32 at a time. S = Q K^T via WMMA (K=64
// chained over two k=32 chunks), online softmax, O += P V via WMMA.
// q/k/v/o layout: [B, T, H*HS] row-major (head-concatenated), f16.
// ---------------------------------------------------------------------------
__global__ __launch_bounds__(128) void attn_kernel(
    const u16* __restrict__ qbuf, const u16* __restrict__ kbuf,
    const u16* __restrict__ vbuf, u16* __restrict__ obuf)
{
    __shared__ u16 Vsh[64][32];        // [feature][key] for this 32-key tile
    __shared__ u16 Psh[4][16][32];     // per-wave P repack buffer

    const int tid  = threadIdx.x;
    const int wave = tid >> 5;
    const int lane = tid & 31;
    const int half = lane >> 4;
    const int l16  = lane & 15;
    const int bb = blockIdx.z, hh = blockIdx.y;
    const int qbase = blockIdx.x * 64;
    const float scale = 0.125f;        // 1/sqrt(HS)

    // Q fragments (rows qbase + wave*16 + l16, all 64 features)
    AFrag qa[2];
    {
        const int qg = qbase + wave * 16 + l16;
        const u16* qp = qbuf + ((long)(bb * GPT_T) + qg) * GPT_D + hh * GPT_HS;
#pragma unroll
        for (int c = 0; c < 2; ++c) {
            qa[c].q[0] = *(const uint4*)(qp + c * 32 + half * 8);
            qa[c].q[1] = *(const uint4*)(qp + c * 32 + 16 + half * 8);
        }
    }

    float m_i[8], l_i[8];
    CFrag Oacc[4];
#pragma unroll
    for (int r = 0; r < 8; ++r) { m_i[r] = -3e38f; l_i[r] = 0.f; }
#pragma unroll
    for (int n = 0; n < 4; ++n) {
#pragma unroll
        for (int r = 0; r < 8; ++r) Oacc[n].f[r] = 0.f;
    }

    const int send = (qbase + 63) / 32;      // inclusive key-tile bound
    for (int s = 0; s <= send; ++s) {
        __syncthreads();
        // stage V tile [32 keys x 64 feats] transposed into Vsh[feat][key]
#pragma unroll
        for (int i = 0; i < 2; ++i) {
            const int li  = tid + i * 128;
            const int key = li >> 3;
            const int fc  = li & 7;
            uint4 d = *(const uint4*)&vbuf[((long)(bb * GPT_T) + s * 32 + key) * GPT_D
                                           + hh * GPT_HS + fc * 8];
            const u16* ds = (const u16*)&d;
#pragma unroll
            for (int j = 0; j < 8; ++j) Vsh[fc * 8 + j][key] = ds[j];
        }
        __syncthreads();

        // scores for the two 16-key subtiles
        CFrag S[2];
#pragma unroll
        for (int sub = 0; sub < 2; ++sub) {
#pragma unroll
            for (int r = 0; r < 8; ++r) S[sub].f[r] = 0.f;
#pragma unroll
            for (int c = 0; c < 2; ++c) {
                AFrag kf;   // B-frag of K^T: col = key, k = feature
                const int key = s * 32 + sub * 16 + l16;
                const u16* kp = kbuf + ((long)(bb * GPT_T) + key) * GPT_D
                                + hh * GPT_HS + c * 32 + half * 8;
                kf.q[0] = *(const uint4*)kp;
                kf.q[1] = *(const uint4*)(kp + 16);
                S[sub].v = __builtin_amdgcn_wmma_f32_16x16x32_f16(
                    false, qa[c].v, false, kf.v, (short)0, S[sub].v, false, false);
            }
        }

        // online softmax over the 32 keys of this tile
#pragma unroll
        for (int r = 0; r < 8; ++r) {
            const int row = qbase + wave * 16 + r + 8 * half;   // global query
            float s0 = S[0].f[r] * scale;
            float s1 = S[1].f[r] * scale;
            const int k0 = s * 32 + l16;
            const int k1 = s * 32 + 16 + l16;
            if (k0 > row) s0 = -1e30f;
            if (k1 > row) s1 = -1e30f;
            float mx = fmaxf(s0, s1);
#pragma unroll
            for (int off = 1; off < 16; off <<= 1)
                mx = fmaxf(mx, __shfl_xor(mx, off, 32));
            const float mnew = fmaxf(m_i[r], mx);
            const float p0 = expf(s0 - mnew);
            const float p1 = expf(s1 - mnew);
            float ps = p0 + p1;
#pragma unroll
            for (int off = 1; off < 16; off <<= 1)
                ps += __shfl_xor(ps, off, 32);
            const float alpha = expf(m_i[r] - mnew);
            l_i[r] = l_i[r] * alpha + ps;
            m_i[r] = mnew;
#pragma unroll
            for (int n = 0; n < 4; ++n) Oacc[n].f[r] *= alpha;
            Psh[wave][r + 8 * half][l16]      = f2h(p0);
            Psh[wave][r + 8 * half][16 + l16] = f2h(p1);
        }
        __builtin_amdgcn_wave_barrier();   // order per-wave LDS P write->read

        // repack P as A-frag and accumulate O += P @ V
        AFrag pa;
        {
            const u16* pp = &Psh[wave][l16][half * 8];
            pa.q[0] = *(const uint4*)pp;
            pa.q[1] = *(const uint4*)(pp + 16);
        }
#pragma unroll
        for (int n = 0; n < 4; ++n) {
            AFrag vf;   // B-frag of V: col = feature, k = key
            const u16* vp = &Vsh[n * 16 + l16][half * 8];
            vf.q[0] = *(const uint4*)vp;
            vf.q[1] = *(const uint4*)(vp + 16);
            Oacc[n].v = __builtin_amdgcn_wmma_f32_16x16x32_f16(
                false, pa.v, false, vf.v, (short)0, Oacc[n].v, false, false);
        }
    }

    // finalize and store head output (f16, head-concatenated layout)
#pragma unroll
    for (int n = 0; n < 4; ++n) {
#pragma unroll
        for (int r = 0; r < 8; ++r) {
            const int row = qbase + wave * 16 + r + 8 * half;
            const float val = Oacc[n].f[r] / l_i[r];
            obuf[((long)(bb * GPT_T) + row) * GPT_D + hh * GPT_HS + n * 16 + l16]
                = f2h(val);
        }
    }
}

// ---------------------------------------------------------------------------
// Host-side orchestration (all launches on `stream`; deterministic)
// ---------------------------------------------------------------------------
extern "C" void kernel_launch(void* const* d_in, const int* in_sizes, int n_in,
                              void* d_out, int out_size, void* d_ws, size_t ws_size,
                              hipStream_t stream) {
    const int*   idx     = (const int*)  d_in[0];
    const float* tok_emb = (const float*)d_in[1];
    const float* out_b   = (const float*)d_in[2];
    const float* Wq      = (const float*)d_in[3];
    const float* Wk      = (const float*)d_in[4];
    const float* Wv      = (const float*)d_in[5];
    const float* Wp      = (const float*)d_in[6];
    const float* bp      = (const float*)d_in[7];
    const float* W1      = (const float*)d_in[8];
    const float* b1      = (const float*)d_in[9];
    const float* W2      = (const float*)d_in[10];
    const float* b2      = (const float*)d_in[11];
    const float* ln1_g   = (const float*)d_in[12];
    const float* ln1_b   = (const float*)d_in[13];
    const float* ln2_g   = (const float*)d_in[14];
    const float* ln2_b   = (const float*)d_in[15];
    const float* lnf_g   = (const float*)d_in[16];
    const float* lnf_b   = (const float*)d_in[17];
    float* out = (float*)d_out;

    // workspace carve-up (~44 MB total)
    char* ws = (char*)d_ws;
    float* x   = (float*)ws;                  ws += (size_t)GPT_M * GPT_D * 4;   // residual fp32
    u16*   h   = (u16*)ws;                    ws += (size_t)GPT_M * GPT_D * 2;   // LN out f16
    u16*   qb  = (u16*)ws;                    ws += (size_t)GPT_M * GPT_D * 2;
    u16*   kb  = (u16*)ws;                    ws += (size_t)GPT_M * GPT_D * 2;
    u16*   vb  = (u16*)ws;                    ws += (size_t)GPT_M * GPT_D * 2;
    u16*   ob  = (u16*)ws;                    ws += (size_t)GPT_M * GPT_D * 2;
    u16*   hid = (u16*)ws;                    ws += (size_t)GPT_M * GPT_FF * 2;

    embed_kernel<<<GPT_M, 256, 0, stream>>>(idx, tok_emb, x);

    const dim3 gqkv(1, GPT_M / 128, GPT_H);
    const dim3 gproj(GPT_D / 128, GPT_M / 128, 1);
    const dim3 gfc1(GPT_FF / 128, GPT_M / 128, 1);
    const dim3 gattn(GPT_T / 64, GPT_H, GPT_B);

    for (int l = 0; l < GPT_L; ++l) {
        ln_kernel<<<GPT_M, 256, 0, stream>>>(x, ln1_g + (long)l * GPT_D,
                                             ln1_b + (long)l * GPT_D, h);
        const long wsz = (long)GPT_H * GPT_D * GPT_HS;
        // q/k/v projections, one head per blockIdx.z (N = HS = 64)
        gemm_kernel<64, false, false, false, false, false, true><<<gqkv, 256, 0, stream>>>(
            h, GPT_D, Wq + (long)l * wsz, GPT_HS, (long)GPT_D * GPT_HS,
            nullptr, nullptr, 0, nullptr, 0, qb, GPT_D, GPT_D, GPT_HS);
        gemm_kernel<64, false, false, false, false, false, true><<<gqkv, 256, 0, stream>>>(
            h, GPT_D, Wk + (long)l * wsz, GPT_HS, (long)GPT_D * GPT_HS,
            nullptr, nullptr, 0, nullptr, 0, kb, GPT_D, GPT_D, GPT_HS);
        gemm_kernel<64, false, false, false, false, false, true><<<gqkv, 256, 0, stream>>>(
            h, GPT_D, Wv + (long)l * wsz, GPT_HS, (long)GPT_D * GPT_HS,
            nullptr, nullptr, 0, nullptr, 0, vb, GPT_D, GPT_D, GPT_HS);

        attn_kernel<<<gattn, 128, 0, stream>>>(qb, kb, vb, ob);

        // x = x + o @ Wp + bp
        gemm_kernel<128, false, true, false, true, true, false><<<gproj, 256, 0, stream>>>(
            ob, GPT_D, Wp + (long)l * GPT_D * GPT_D, GPT_D, 0,
            bp + (long)l * GPT_D, x, GPT_D, x, GPT_D, nullptr, 0, GPT_D, 0);

        ln_kernel<<<GPT_M, 256, 0, stream>>>(x, ln2_g + (long)l * GPT_D,
                                             ln2_b + (long)l * GPT_D, h);
        // hid = relu(h @ W1 + b1)
        gemm_kernel<128, false, true, true, false, false, true><<<gfc1, 256, 0, stream>>>(
            h, GPT_D, W1 + (long)l * GPT_D * GPT_FF, GPT_FF, 0,
            b1 + (long)l * GPT_FF, nullptr, 0, nullptr, 0, hid, GPT_FF, GPT_D, 0);
        // x = x + hid @ W2 + b2
        gemm_kernel<128, false, true, false, true, true, false><<<gproj, 256, 0, stream>>>(
            hid, GPT_FF, W2 + (long)l * GPT_FF * GPT_D, GPT_D, 0,
            b2 + (long)l * GPT_D, x, GPT_D, x, GPT_D, nullptr, 0, GPT_FF, 0);
    }

    ln_kernel<<<GPT_M, 256, 0, stream>>>(x, lnf_g, lnf_b, h);

    // logits = h @ tok_emb^T + out_b  (transposed-B read of the tied embedding)
    const dim3 glog(GPT_V / 128, GPT_M / 128, 1);
    gemm_kernel<128, true, true, false, false, true, false><<<glog, 256, 0, stream>>>(
        h, GPT_D, tok_emb, GPT_D, 0,
        out_b, nullptr, 0, out, GPT_V, nullptr, 0, GPT_D, 0);
}
